// LayerStacks_85392539779291
// MI455X (gfx1250) — matrix operand order repaired and checked by reading, not compiled, not run on hardware
//
#include <hip/hip_runtime.h>
#include <hip/hip_bf16.h>

typedef __attribute__((ext_vector_type(16))) _Float16 v16h;
typedef __attribute__((ext_vector_type(8)))  _Float16 v8h;
typedef __attribute__((ext_vector_type(8)))  float    v8f;
typedef __attribute__((ext_vector_type(4)))  float    v4f;

#define L1N   3072
#define COUNT 8
#define BATCH 8192
#define MAX_TILES 528
#define KSPLIT 8          // waves per block (wave32)
#define KPW   (L1N / KSPLIT)   // 384 K per wave
#define WMMA_K 32
#define PF_DIST (4 * WMMA_K)   // prefetch 512B/lane ahead of the pipelined loads

// ---------------- prep: fold shared weights, convert to f16, pad w2 ----------------
__global__ void ls_prep(const float* __restrict__ w1_stack,
                        const float* __restrict__ b1_stack,
                        const float* __restrict__ w1_shared,
                        const float* __restrict__ b1_shared,
                        const float* __restrict__ w2,
                        _Float16* __restrict__ w1c,
                        float* __restrict__ b1c,
                        _Float16* __restrict__ w2p)
{
    const int N1 = COUNT * 16 * L1N;   // 393216
    const int N2 = COUNT * 32 * 32;    // 8192 (padded K=32)
    int e = blockIdx.x * blockDim.x + threadIdx.x;
    if (e < N1) {
        int rem = e % (16 * L1N);
        w1c[e] = (_Float16)(w1_stack[e] + w1_shared[rem]);
    } else if (e < N1 + N2) {
        int i = e - N1;
        int s = i >> 10;           // stack
        int r = i & 1023;
        int m = r >> 5;            // output neuron 0..31
        int k = r & 31;            // padded K
        w2p[i] = (k < 30) ? (_Float16)w2[(s * 32 + m) * 30 + k] : (_Float16)0.0f;
    } else if (e < N1 + N2 + COUNT * 16) {
        int i = e - N1 - N2;
        b1c[i] = b1_stack[i] + b1_shared[i & 15];
    }
}

// ---------------- bucketing by ls_indices ----------------
__global__ void ls_zero(int* counts) { if (threadIdx.x < COUNT) counts[threadIdx.x] = 0; }

__global__ void ls_hist(const int* __restrict__ idx, int* __restrict__ counts) {
    int b = blockIdx.x * blockDim.x + threadIdx.x;
    if (b < BATCH) atomicAdd(&counts[idx[b] & (COUNT - 1)], 1);
}

__global__ void ls_plan(const int* __restrict__ counts, int* __restrict__ cursors,
                        int* __restrict__ tBucket, int* __restrict__ tStart,
                        int* __restrict__ tRows) {
    if (blockIdx.x == 0 && threadIdx.x == 0) {
        int startv[COUNT]; int acc = 0;
        for (int i = 0; i < COUNT; ++i) { startv[i] = acc; cursors[i] = acc; acc += counts[i]; }
        int t = 0;
        for (int i = 0; i < COUNT; ++i) {
            int c = counts[i], p = startv[i];
            while (c > 0 && t < MAX_TILES) {
                tBucket[t] = i; tStart[t] = p; tRows[t] = (c < 16) ? c : 16;
                p += 16; c -= 16; ++t;
            }
        }
        for (; t < MAX_TILES; ++t) { tBucket[t] = 0; tStart[t] = 0; tRows[t] = 0; }
    }
}

__global__ void ls_scatter(const int* __restrict__ idx, int* __restrict__ cursors,
                           int* __restrict__ perm) {
    int b = blockIdx.x * blockDim.x + threadIdx.x;
    if (b < BATCH) {
        int pos = atomicAdd(&cursors[idx[b] & (COUNT - 1)], 1);
        perm[pos] = b;
    }
}

// ---------------- main: WMMA layer stack over one 16-row bucket tile ----------------
__global__ __launch_bounds__(256) void ls_main(
    const float* __restrict__ x,
    const _Float16* __restrict__ w1c,
    const float* __restrict__ b1c,
    const _Float16* __restrict__ w2p,
    const float* __restrict__ b2,
    const float* __restrict__ w3,
    const float* __restrict__ b3,
    const int* __restrict__ perm,
    const int* __restrict__ tBucket,
    const int* __restrict__ tStart,
    const int* __restrict__ tRows,
    float* __restrict__ out)
{
    __shared__ float partial[KSPLIT][256];          // per-wave 16x16 f32 partials
    __shared__ __align__(32) _Float16 aLds[16][32]; // activations: [batch col n][k]
    __shared__ float l2Lds[32][16];                 // clipped layer-2: [o][n]
    __shared__ float l1out[16];                     // raw residual (L1 output 15)
    __shared__ int   rrow[16];                      // batch row per tile column

    const int t = blockIdx.x;
    const int rows = tRows[t];
    if (rows == 0) return;                          // uniform per block
    const int bucket = tBucket[t];
    const int start  = tStart[t];

    const int tid  = threadIdx.x;
    const int wave = tid >> 5;
    const int lane = tid & 31;
    const int half = lane >> 4;                     // K-half selector (wave32 layout)
    const int nl   = lane & 15;                     // matrix row/col within 16-tile

    if (tid < 16) {
        int n = (tid < rows) ? tid : 0;             // clamp dead columns to a valid row
        rrow[tid] = perm[start + n];
    }
    __syncthreads();

    // ---- layer 1: D(16x16) += W1c(16x3072,f16) x X^T(3072x16), K split 8 ways ----
    const _Float16* wrow = w1c + ((size_t)bucket * 16 + nl) * L1N;  // A row m = nl
    const float*    xrow = x + (size_t)rrow[nl] * L1N;              // B col n = nl
    const int kbase = wave * KPW;
    v8f acc = {};
    for (int kk = 0; kk < KPW; kk += WMMA_K) {
        const int k0 = kbase + kk;
        if (kk + PF_DIST < KPW)                     // WGP-scope prefetch, 512B/lane ahead
            __builtin_prefetch(xrow + k0 + PF_DIST + half * 16, 0, 3);
        // A fragment: lanes 0-15 hold K 0..15, lanes 16-31 K 16..31
        const _Float16* wp = wrow + k0 + half * 16;
        v8h alo = *(const v8h*)(wp);
        v8h ahi = *(const v8h*)(wp + 8);
        v16h a;
        #pragma unroll
        for (int i = 0; i < 8; ++i) { a[i] = alo[i]; a[8 + i] = ahi[i]; }
        // B fragment: lanes 0-15 hold K 0..15, lanes 16-31 K 16..31 (col = lane&15)
        // x is streamed exactly once -> non-temporal so it never evicts the hot weights
        const v4f* xp4 = (const v4f*)(xrow + k0 + half * 16);
        v4f x0 = __builtin_nontemporal_load(xp4 + 0);
        v4f x1 = __builtin_nontemporal_load(xp4 + 1);
        v4f x2 = __builtin_nontemporal_load(xp4 + 2);
        v4f x3 = __builtin_nontemporal_load(xp4 + 3);
        v16h b;
        #pragma unroll
        for (int i = 0; i < 4; ++i) {
            b[i]      = (_Float16)x0[i];
            b[4 + i]  = (_Float16)x1[i];
            b[8 + i]  = (_Float16)x2[i];
            b[12 + i] = (_Float16)x3[i];
        }
        acc = __builtin_amdgcn_wmma_f32_16x16x32_f16(false, a, false, b,
                                                     (short)0, acc, false, false);
    }
    #pragma unroll
    for (int j = 0; j < 8; ++j) partial[wave][lane * 8 + j] = acc[j];
    __syncthreads();

    // ---- cross-wave K reduction + bias + activation ----
    {
        float s = 0.0f;
        #pragma unroll
        for (int w = 0; w < KSPLIT; ++w) s += partial[w][tid];
        // element (tid) maps to: lane' = tid>>3, j = tid&7 -> o = j + 8*(lane'>=16), n = lane'&15
        int el = tid >> 3, j = tid & 7;
        int o  = j + ((el >> 4) << 3);
        int n  = el & 15;
        float v = s + b1c[bucket * 16 + o];
        if (o < 15) {
            float sq = v * v * (127.0f / 128.0f);
            aLds[n][o]      = (_Float16)fminf(fmaxf(sq, 0.0f), 1.0f); // squared branch k=0..14
            aLds[n][15 + o] = (_Float16)fminf(fmaxf(v,  0.0f), 1.0f); // linear  branch k=15..29
        } else { // o == 15: raw residual + pad K
            l1out[n] = v;
            aLds[n][30] = (_Float16)0.0f;
            aLds[n][31] = (_Float16)0.0f;
        }
    }
    __syncthreads();

    // ---- layer 2: (32x32 padded, f16) x act(32x16) as two 16x16x32 WMMAs ----
    if (wave < 2) {
        const int mtile = wave;
        const _Float16* w2row = w2p + (((size_t)bucket * 32 + mtile * 16 + nl) * 32) + half * 16;
        v8h alo = *(const v8h*)(w2row);
        v8h ahi = *(const v8h*)(w2row + 8);
        v16h a;
        #pragma unroll
        for (int i = 0; i < 8; ++i) { a[i] = alo[i]; a[8 + i] = ahi[i]; }
        v16h b = *(const v16h*)(&aLds[nl][half * 16]);
        v8f c;
        #pragma unroll
        for (int j = 0; j < 8; ++j) c[j] = b2[bucket * 32 + mtile * 16 + half * 8 + j]; // bias via C
        c = __builtin_amdgcn_wmma_f32_16x16x32_f16(false, a, false, b,
                                                   (short)0, c, false, false);
        #pragma unroll
        for (int j = 0; j < 8; ++j) {
            int o = mtile * 16 + half * 8 + j;
            l2Lds[o][nl] = fminf(fmaxf(c[j], 0.0f), 1.0f);
        }
    }
    __syncthreads();

    // ---- layer 3 (1x32 dot) + residual ----
    if (tid < 16 && tid < rows) {
        float s = b3[bucket];
        #pragma unroll
        for (int o = 0; o < 32; ++o) s += w3[bucket * 32 + o] * l2Lds[o][tid];
        __builtin_nontemporal_store(s + l1out[tid], &out[perm[start + tid]]);
    }
}

// ---------------- launch ----------------
extern "C" void kernel_launch(void* const* d_in, const int* in_sizes, int n_in,
                              void* d_out, int out_size, void* d_ws, size_t ws_size,
                              hipStream_t stream) {
    const float* x         = (const float*)d_in[0];
    const int*   ls_idx    = (const int*)  d_in[1];
    const float* w1_stack  = (const float*)d_in[2];
    const float* b1_stack  = (const float*)d_in[3];
    const float* w1_shared = (const float*)d_in[4];
    const float* b1_shared = (const float*)d_in[5];
    const float* w2        = (const float*)d_in[6];
    const float* b2        = (const float*)d_in[7];
    const float* w3        = (const float*)d_in[8];
    const float* b3        = (const float*)d_in[9];
    float* out = (float*)d_out;

    // workspace layout (256B-aligned chunks)
    char* ws = (char*)d_ws;
    _Float16* w1c    = (_Float16*)(ws);                       // 786432 B
    _Float16* w2p    = (_Float16*)(ws + 786432);              // 16384 B
    float*    b1c    = (float*)   (ws + 786432 + 16384);      // 512 B
    int*      counts = (int*)     (ws + 803584);              // 32 B
    int*      cursors= (int*)     (ws + 803840);              // 32 B
    int*      perm   = (int*)     (ws + 804096);              // 32768 B
    int*      tBkt   = (int*)     (ws + 836864);              // 2112 B
    int*      tSta   = (int*)     (ws + 839168);
    int*      tRow   = (int*)     (ws + 841472);

    const int prepN = COUNT * 16 * L1N + COUNT * 32 * 32 + COUNT * 16;
    ls_prep<<<(prepN + 255) / 256, 256, 0, stream>>>(w1_stack, b1_stack, w1_shared,
                                                     b1_shared, w2, w1c, b1c, w2p);
    ls_zero<<<1, 32, 0, stream>>>(counts);
    ls_hist<<<BATCH / 256, 256, 0, stream>>>(ls_idx, counts);
    ls_plan<<<1, 32, 0, stream>>>(counts, cursors, tBkt, tSta, tRow);
    ls_scatter<<<BATCH / 256, 256, 0, stream>>>(ls_idx, cursors, perm);
    ls_main<<<MAX_TILES, 256, 0, stream>>>(x, w1c, b1c, w2p, b2, w3, b3,
                                           perm, tBkt, tSta, tRow, out);
    (void)in_sizes; (void)n_in; (void)out_size; (void)ws_size;
}
